// MultiheadAttention_33578054320633
// MI455X (gfx1250) — matrix operand order repaired and checked by reading, not compiled, not run on hardware
//
#include <hip/hip_runtime.h>
#include <hip/hip_bf16.h>

// ---------------------------------------------------------------------------
// Causal multi-head attention, bf16 WMMA pipeline for gfx1250 (MI455X).
//   B=2, L=2048, D=1024, H=16, hd=64.  All matmuls via v_wmma_f32_16x16x32_bf16.
//   GEMMs: 128x128 block tile, 4 waves x (64x64) wave tiles; tiles staged into
//   LDS with GLOBAL_LOAD_ASYNC_TO_LDS_B128 (ASYNCcnt) when available.
// ---------------------------------------------------------------------------

typedef __bf16 bf16;
typedef __attribute__((ext_vector_type(4)))  __bf16 v4bf;
typedef __attribute__((ext_vector_type(8)))  __bf16 v8bf;
typedef __attribute__((ext_vector_type(16))) __bf16 v16bf;
typedef __attribute__((ext_vector_type(8)))  float  v8f;
typedef __attribute__((ext_vector_type(4)))  int    v4i;

static constexpr int Bb = 2;
static constexpr int Ls = 2048;
static constexpr int Dd = 1024;
static constexpr int Hh = 16;
static constexpr int HD = 64;           // head dim
static constexpr int Mtok = Bb * Ls;    // 4096 rows

#define WMMA_BF16(a, b, c) \
  __builtin_amdgcn_wmma_f32_16x16x32_bf16(false, (a), false, (b), (short)0, (c), false, false)

// ---- gfx1250 async global->LDS staging support ----------------------------

#if defined(__has_builtin)
#if __has_builtin(__builtin_amdgcn_global_load_async_to_lds_b128)
#define HAVE_ASYNC_LDS 1
#endif
#if __has_builtin(__builtin_amdgcn_s_wait_asynccnt)
#define WAIT_ASYNC() __builtin_amdgcn_s_wait_asynccnt(0)
#endif
#endif
#ifndef HAVE_ASYNC_LDS
#define HAVE_ASYNC_LDS 0
#endif
#ifndef WAIT_ASYNC
#define WAIT_ASYNC() asm volatile("s_wait_asynccnt 0" ::: "memory")
#endif

// pointer-to-vector types in explicit address spaces (per builtin signature:
// param0 = v4i addrspace(1)* global source, param1 = v4i addrspace(3)* LDS dst)
typedef __attribute__((address_space(1))) v4i* gv4i_p;
typedef __attribute__((address_space(3))) v4i* lv4i_p;

// ---- fragment loaders (contiguous k order, same for A and B) -----------------------

__device__ __forceinline__ v16bf cat16(v8bf lo, v8bf hi) {
  v16bf r;
#pragma unroll
  for (int i = 0; i < 8; ++i) { r[i] = lo[i]; r[i + 8] = hi[i]; }
  return r;
}

// A-matrix 16x32 bf16: row = lane%16; lane<16 -> K 0..15, lane>=16 -> K 16..31 (contiguous)
__device__ __forceinline__ v16bf load_a_frag(const bf16* rowp, int lane) {
  const int hb = (lane & 16) ? 16 : 0;
  v8bf c0 = *(const v8bf*)(rowp + hb);
  v8bf c1 = *(const v8bf*)(rowp + hb + 8);
  return cat16(c0, c1);
}

// B-matrix 32x16 bf16: col = lane%16; lane<16 -> K 0..15, lane>=16 -> K 16..31 (contiguous)
__device__ __forceinline__ v16bf load_b_frag(const bf16* colp, int lane) {
  const int hb = (lane & 16) ? 16 : 0;
  v8bf c0 = *(const v8bf*)(colp + hb);
  v8bf c1 = *(const v8bf*)(colp + hb + 8);
  return cat16(c0, c1);
}

// ---- fp32 -> bf16 conversion ----------------------------------------------

__global__ void f32_to_bf16_kernel(const float* __restrict__ in,
                                   bf16* __restrict__ out, size_t n4) {
  size_t i = (size_t)blockIdx.x * blockDim.x + threadIdx.x;
  size_t stride = (size_t)gridDim.x * blockDim.x;
  const float4* in4 = (const float4*)in;
  v4bf* out4 = (v4bf*)out;
  for (; i < n4; i += stride) {
    float4 f = in4[i];
    v4bf o;
    o[0] = (bf16)f.x; o[1] = (bf16)f.y; o[2] = (bf16)f.z; o[3] = (bf16)f.w;
    out4[i] = o;
  }
}

// ---- tiled WMMA GEMM: out = A[M,K] @ W[N,K]^T -----------------------------
// Block: 128 threads = 4 waves, block tile 128(M) x 128(N), wave tile 64x64.
// K loop step 32, tiles staged into LDS (async when available).
// LAYOUT 0: float out [M,N] row-major (final projection)
// LAYOUT 1: bf16 out  [B,H,L,hd]    (Q, K)
// LAYOUT 2: bf16 out  [B,H,hd,L]    (V transposed)

template <int LAYOUT>
__global__ __launch_bounds__(128) void gemm_bf16_wmma(const bf16* __restrict__ A,
                                                      const bf16* __restrict__ W,
                                                      void* __restrict__ outv,
                                                      int M, int N, int K) {
  __shared__ bf16 lA[128 * 32];   // 8 KB
  __shared__ bf16 lB[128 * 32];   // 8 KB

  const int tid  = threadIdx.x;
  const int lane = tid & 31;
  const int wave = tid >> 5;
  const int m0 = blockIdx.y * 128;
  const int n0 = blockIdx.x * 128;
  const int mw = (wave >> 1) * 64;   // wave tile origin inside block tile
  const int nw = (wave & 1) * 64;

  v8f acc[4][4] = {};

  for (int k0 = 0; k0 < K; k0 += 32) {
    // ---- global -> LDS staging: 512 x 16B chunks per tile ----
#pragma unroll
    for (int it = 0; it < 4; ++it) {
      const int c   = tid + it * 128;      // chunk id 0..511
      const int row = c >> 2;              // 0..127
      const int off = (c & 3) * 8;         // 0,8,16,24
      const bf16* ga = A + (size_t)(m0 + row) * K + k0 + off;
      const bf16* gb = W + (size_t)(n0 + row) * K + k0 + off;
      bf16* la = &lA[row * 32 + off];
      bf16* lb = &lB[row * 32 + off];
#if HAVE_ASYNC_LDS
      __builtin_amdgcn_global_load_async_to_lds_b128(
          (gv4i_p)(const void*)ga, (lv4i_p)(void*)la, 0, 0);
      __builtin_amdgcn_global_load_async_to_lds_b128(
          (gv4i_p)(const void*)gb, (lv4i_p)(void*)lb, 0, 0);
#else
      *(v8bf*)la = *(const v8bf*)ga;
      *(v8bf*)lb = *(const v8bf*)gb;
      if (k0 + 32 < K) {                   // prefetch next K tile into caches
        __builtin_prefetch(ga + 32, 0, 1);
        __builtin_prefetch(gb + 32, 0, 1);
      }
#endif
    }
#if HAVE_ASYNC_LDS
    WAIT_ASYNC();
#endif
    __syncthreads();

    // ---- fragments from LDS, 16 WMMAs per K-step ----
    v16bf a[4], b[4];
#pragma unroll
    for (int i = 0; i < 4; ++i)
      a[i] = load_a_frag(&lA[(mw + i * 16 + (lane & 15)) * 32], lane);
#pragma unroll
    for (int j = 0; j < 4; ++j)
      b[j] = load_b_frag(&lB[(nw + j * 16 + (lane & 15)) * 32], lane);
#pragma unroll
    for (int i = 0; i < 4; ++i)
#pragma unroll
      for (int j = 0; j < 4; ++j)
        acc[i][j] = WMMA_BF16(a[i], b[j], acc[i][j]);
    __syncthreads();
  }

  // ---- epilogue ----
  const int rsel = (lane & 16) ? 8 : 0;
#pragma unroll
  for (int i = 0; i < 4; ++i)
#pragma unroll
    for (int j = 0; j < 4; ++j)
#pragma unroll
      for (int r = 0; r < 8; ++r) {
        const int row = m0 + mw + i * 16 + r + rsel;
        const int col = n0 + nw + j * 16 + (lane & 15);
        const float v = acc[i][j][r];
        if (LAYOUT == 0) {
          ((float*)outv)[(size_t)row * N + col] = v;
        } else {
          const int bb = row / Ls, ll = row % Ls;
          const int hh = col / HD, tt = col % HD;
          if (LAYOUT == 1)
            ((bf16*)outv)[(((size_t)(bb * Hh + hh)) * Ls + ll) * HD + tt] = (bf16)v;
          else
            ((bf16*)outv)[(((size_t)(bb * Hh + hh)) * HD + tt) * Ls + ll] = (bf16)v;
        }
      }
}

// ---- flash attention: one wave per 16-query tile per (b,h) ----------------
// Q,K: [B,H,L,hd] bf16; Vt: [B,H,hd,L] bf16; ctx: [B*L, D] bf16.

__global__ __launch_bounds__(256) void flash_attn_wmma(const bf16* __restrict__ Qh,
                                                       const bf16* __restrict__ Kh,
                                                       const bf16* __restrict__ Vt,
                                                       bf16* __restrict__ ctx) {
  __shared__ bf16 smem[8 * 16 * 32];   // one 16x32 P tile per wave
  const int lane = threadIdx.x & 31;
  const int wib  = threadIdx.x >> 5;
  bf16* pl = smem + wib * 512;

  const int wid = blockIdx.x * 8 + wib;          // 0..4095
  const int qt  = wid & (Ls / 16 - 1);           // query tile in sequence
  const int bh  = wid / (Ls / 16);               // 0..31
  const int q0  = qt * 16;

  const bf16* Qp = Qh + (size_t)bh * Ls * HD;
  const bf16* Kp = Kh + (size_t)bh * Ls * HD;
  const bf16* Vp = Vt + (size_t)bh * HD * Ls;

  // Q fragments (16 rows x 64 K) -> two 16x32 A fragments, loaded once
  const bf16* qrow = Qp + (size_t)(q0 + (lane & 15)) * HD;
  const v16bf qa0 = load_a_frag(qrow, lane);
  const v16bf qa1 = load_a_frag(qrow + 32, lane);

  float mrow[8], lrow[8];
  v8f o[4] = {};
#pragma unroll
  for (int r = 0; r < 8; ++r) { mrow[r] = -3.0e38f; lrow[r] = 0.0f; }

  const float SCALE = 0.125f;                     // 1/sqrt(64)
  const int rbase = q0 + ((lane & 16) ? 8 : 0);

  for (int j0 = 0; j0 < q0 + 16; j0 += 32) {
    // --- S = Q @ K^T for 32 keys (two 16-col N tiles) ---
    const bf16* kc0 = Kp + (size_t)(j0 + (lane & 15)) * HD;
    const bf16* kc1 = Kp + (size_t)(j0 + 16 + (lane & 15)) * HD;
    v8f s0 = {}, s1 = {};
    s0 = WMMA_BF16(qa0, load_b_frag(kc0, lane), s0);
    s0 = WMMA_BF16(qa1, load_b_frag(kc0 + 32, lane), s0);
    s1 = WMMA_BF16(qa0, load_b_frag(kc1, lane), s1);
    s1 = WMMA_BF16(qa1, load_b_frag(kc1 + 32, lane), s1);

    // --- scale + causal mask (only diagonal-adjacent iterations) ---
    const int c0 = j0 + (lane & 15), c1 = c0 + 16;
    const bool partial = (j0 + 31) > q0;
#pragma unroll
    for (int r = 0; r < 8; ++r) {
      float a = s0[r] * SCALE, b = s1[r] * SCALE;
      if (partial) {
        const int row = rbase + r;
        if (c0 > row) a = -3.0e38f;
        if (c1 > row) b = -3.0e38f;
      }
      s0[r] = a; s1[r] = b;
    }

    // --- online softmax (row stats per lane-half, matching C/D layout) ---
#pragma unroll
    for (int r = 0; r < 8; ++r) {
      float v = fmaxf(s0[r], s1[r]);
#pragma unroll
      for (int off = 1; off < 16; off <<= 1) v = fmaxf(v, __shfl_xor(v, off, 32));
      const float mn = fmaxf(mrow[r], v);
      const float alpha = __expf(mrow[r] - mn);
      mrow[r] = mn;
      const float p0 = __expf(s0[r] - mn);
      const float p1 = __expf(s1[r] - mn);
      s0[r] = p0; s1[r] = p1;
      float sum = p0 + p1;
#pragma unroll
      for (int off = 1; off < 16; off <<= 1) sum += __shfl_xor(sum, off, 32);
      lrow[r] = lrow[r] * alpha + sum;
#pragma unroll
      for (int t = 0; t < 4; ++t) o[t][r] *= alpha;
    }

    // --- transpose P (D-layout -> A-layout) through LDS ---
    {
      const int row = (lane & 16) ? 8 : 0;
      const int col = lane & 15;
#pragma unroll
      for (int r = 0; r < 8; ++r) {
        pl[(row + r) * 32 + col]      = (bf16)s0[r];
        pl[(row + r) * 32 + col + 16] = (bf16)s1[r];
      }
    }
    asm volatile("s_wait_dscnt 0" ::: "memory");
    const v16bf pa = load_a_frag(pl + (lane & 15) * 32, lane);

    // --- O += P @ V  (V^T layout -> contiguous B fragments) ---
#pragma unroll
    for (int t = 0; t < 4; ++t) {
      const bf16* vcol = Vp + (size_t)(t * 16 + (lane & 15)) * Ls + j0;
      o[t] = WMMA_BF16(pa, load_b_frag(vcol, lane), o[t]);
    }
  }

  // --- normalize and write context (concat-head layout [B*L, D]) ---
  const int bb = bh >> 4, hh = bh & 15;
#pragma unroll
  for (int r = 0; r < 8; ++r) {
    const float inv = 1.0f / lrow[r];
    const int row = rbase + r;
#pragma unroll
    for (int t = 0; t < 4; ++t) {
      const size_t idx =
          ((size_t)(bb * Ls + row)) * Dd + hh * HD + t * 16 + (lane & 15);
      ctx[idx] = (bf16)(o[t][r] * inv);
    }
  }
}

// ---------------------------------------------------------------------------

extern "C" void kernel_launch(void* const* d_in, const int* in_sizes, int n_in,
                              void* d_out, int out_size, void* d_ws, size_t ws_size,
                              hipStream_t stream) {
  const float* x  = (const float*)d_in[0];
  const float* Wq = (const float*)d_in[1];
  const float* Wk = (const float*)d_in[2];
  const float* Wv = (const float*)d_in[3];
  const float* Wo = (const float*)d_in[4];

  const size_t nX = (size_t)Mtok * Dd;   // 4 Mi elements
  const size_t nW = (size_t)Dd * Dd;     // 1 Mi elements

  bf16* xb  = (bf16*)d_ws;
  bf16* wqb = xb + nX;
  bf16* wkb = wqb + nW;
  bf16* wvb = wkb + nW;
  bf16* wob = wvb + nW;
  bf16* Qh  = wob + nW;
  bf16* Kh  = Qh + nX;
  bf16* Vt  = Kh + nX;
  bf16* ctx = Vt + nX;

  // 1) convert inputs to bf16
  f32_to_bf16_kernel<<<2048, 256, 0, stream>>>(x,  xb,  nX / 4);
  f32_to_bf16_kernel<<<1024, 256, 0, stream>>>(Wq, wqb, nW / 4);
  f32_to_bf16_kernel<<<1024, 256, 0, stream>>>(Wk, wkb, nW / 4);
  f32_to_bf16_kernel<<<1024, 256, 0, stream>>>(Wv, wvb, nW / 4);
  f32_to_bf16_kernel<<<1024, 256, 0, stream>>>(Wo, wob, nW / 4);

  // 2) QKV projections (y = x @ W^T), head layouts for attention
  dim3 ggrid(Dd / 128, Mtok / 128);
  gemm_bf16_wmma<1><<<ggrid, 128, 0, stream>>>(xb, wqb, Qh, Mtok, Dd, Dd);
  gemm_bf16_wmma<1><<<ggrid, 128, 0, stream>>>(xb, wkb, Kh, Mtok, Dd, Dd);
  gemm_bf16_wmma<2><<<ggrid, 128, 0, stream>>>(xb, wvb, Vt, Mtok, Dd, Dd);

  // 3) causal flash attention: 4096 waves = 512 blocks x 8 waves
  flash_attn_wmma<<<512, 256, 0, stream>>>(Qh, Kh, Vt, ctx);

  // 4) output projection, fp32 result
  gemm_bf16_wmma<0><<<ggrid, 128, 0, stream>>>(ctx, wob, d_out, Mtok, Dd, Dd);
}
